// RNNModNModel_8418135900414
// MI455X (gfx1250) — compile-verified, matrix operand order checked by reading, not run on hardware
//
#include <hip/hip_runtime.h>
#include <hip/hip_bf16.h>

typedef __attribute__((ext_vector_type(16))) __bf16 v16bf;
typedef __attribute__((ext_vector_type(8)))  float  v8f;

constexpr int kB     = 2048;   // batch
constexpr int kS     = 64;     // encoder steps
constexpr int kH     = 512;    // hidden
constexpr int kSteps = 32;     // decoder steps
constexpr int kBT    = 64;     // batch rows per workgroup (4 WMMA M-tiles)
constexpr int kMT    = 4;      // M-tiles per workgroup
constexpr int kThreads = 512;  // 16 waves
constexpr int kKT    = kH / 32;           // 16 K-tiles of 32
constexpr int kFragLds = kKT * 32 * 16;   // bf16 elems per A-tile (8192)
constexpr int kATile   = kMT * kFragLds;  // bf16 elems per ping-pong buffer

// ---------------------------------------------------------------------------
// Prep: pack Whh [3H, H] (f32 row-major; gh[m,n] = sum_k h[m,k]*Whh[n,k])
// into bf16 WMMA B-fragments.  Fragment (nt, kt): element (lane, e) holds
// B[k, n] with n = nt*16 + lane%16,  k = kt*32 + ((e/8)*16 + (lane/16)*8 + e%8)
// Packed element index = ((nt*16 + kt)*32 + lane)*16 + e   (kt contiguous per nt)
// ---------------------------------------------------------------------------
__global__ void pack_whh_kernel(const float* __restrict__ Whh,
                                __bf16* __restrict__ packed) {
  int tid = blockIdx.x * blockDim.x + threadIdx.x;
  if (tid >= 3 * kH * kH) return;
  int e    = tid & 15;
  int lane = (tid >> 4) & 31;
  int kt   = (tid >> 9) & 15;
  int nt   = tid >> 13;
  int n  = nt * 16 + (lane & 15);
  int kk = ((e >> 3) << 4) | ((lane >> 4) << 3) | (e & 7);
  int k  = kt * 32 + kk;
  packed[tid] = (__bf16)Whh[n * kH + k];
}

__device__ __forceinline__ float sigmoidf_(float x) {
  return 1.0f / (1.0f + __expf(-x));
}

// ---------------------------------------------------------------------------
// Main kernel: 32 blocks x 512 threads (16 waves).  Each block owns 64 batch
// rows (4 M-tiles) through all 96 sequential GRU steps.  Each B fragment
// streamed from L2 feeds 4 WMMAs.  The bf16 A-fragment copy of h is
// ping-pong double-buffered so the gate update writes h immediately (h32 is
// thread-private per element within a step) -> one barrier per step and no
// register-held hnew (no spills).
// ---------------------------------------------------------------------------
__global__ void __launch_bounds__(kThreads)
gru_main_kernel(const float* __restrict__ input_seq,
                const float* __restrict__ Wih,
                const float* __restrict__ bih,
                const float* __restrict__ bhh,
                const float* __restrict__ Wout,
                const float* __restrict__ bout,
                const __bf16* __restrict__ WhhP,
                float* __restrict__ out) {
  __shared__ float h32[kBT * kH];                        // 128 KB f32 master h
  __shared__ __align__(32) __bf16 hbf[2 * kATile];       // 128 KB A-frags (x2)
  __shared__ float x_lds[kBT];
  __shared__ float part[kBT][16];

  const int tid   = threadIdx.x;
  const int lane  = tid & 31;
  const int wave  = tid >> 5;                  // 0..15
  const int bbase = blockIdx.x * kBT;

  // h0 = 0 (only ping-pong buffer 0 needs init; every step rewrites fully)
  for (int i = tid; i < kBT * kH; i += kThreads) h32[i] = 0.0f;
  for (int i = tid; i < kATile; i += kThreads) hbf[i] = (__bf16)0.0f;
  __syncthreads();

  const int jcol  = lane & 15;          // N column within a 16-wide tile
  const int mhalf = (lane >> 4) << 3;   // C/D fragment: m = r + 8*(lane/16)

  const v16bf* Bbase = (const v16bf*)WhhP;   // index: (nt*16 + kt)*32 + lane

  for (int step = 0; step < kS + kSteps; ++step) {
    const bool dec = (step >= kS);
    const int  pp  = step & 1;                              // read buffer
    const v16bf* Ard = (const v16bf*)(hbf + pp * kATile);   // GEMM A source
    __bf16*      Awr = hbf + (1 - pp) * kATile;             // h_new A dest

    if (!dec) {                                   // encoder input
      if (tid < kBT) x_lds[tid] = input_seq[(bbase + tid) * kS + step];
      __syncthreads();
    } else if (step == kS) {                      // first decoder input x0 = 0
      if (tid < kBT) x_lds[tid] = 0.0f;
      __syncthreads();
    } // later decoder steps: x_lds set (with barrier) by previous y-phase

    // ---- GEMM (gh = h @ Whh.T) + gate nonlinearity + immediate h update ---
#pragma unroll
    for (int t = 0; t < 2; ++t) {
      const int jt = wave + t * 16;               // hidden-dim tile 0..31
      v8f accR[kMT], accZ[kMT], accN[kMT];
#pragma unroll
      for (int mt = 0; mt < kMT; ++mt) { accR[mt] = {}; accZ[mt] = {}; accN[mt] = {}; }
      const v16bf* Br = Bbase + (jt     ) * 16 * 32 + lane;  // r gate
      const v16bf* Bz = Bbase + (jt + 32) * 16 * 32 + lane;  // z gate
      const v16bf* Bn = Bbase + (jt + 64) * 16 * 32 + lane;  // n gate
#pragma unroll 4
      for (int kt = 0; kt < kKT; ++kt) {
        v16bf br = Br[kt * 32];
        v16bf bz = Bz[kt * 32];
        v16bf bn = Bn[kt * 32];
#pragma unroll
        for (int mt = 0; mt < kMT; ++mt) {
          v16bf a = Ard[mt * (kFragLds / 16) + kt * 32 + lane];
          accR[mt] = __builtin_amdgcn_wmma_f32_16x16x32_bf16(false, a, false, br,
                                                             (short)0, accR[mt], false, false);
          accZ[mt] = __builtin_amdgcn_wmma_f32_16x16x32_bf16(false, a, false, bz,
                                                             (short)0, accZ[mt], false, false);
          accN[mt] = __builtin_amdgcn_wmma_f32_16x16x32_bf16(false, a, false, bn,
                                                             (short)0, accN[mt], false, false);
        }
      }
      const int j = jt * 16 + jcol;
      const float wir = Wih[j], wiz = Wih[kH + j], win = Wih[2 * kH + j];
      const float brz = bih[j] + bhh[j];
      const float bzz = bih[kH + j] + bhh[kH + j];
      const float bin = bih[2 * kH + j];
      const float bhn = bhh[2 * kH + j];
      // write-index pieces for the bf16 A-fragment layout (constant per t)
      const int ktw = j >> 5, kkw = j & 31;
      const int ew   = ((kkw >> 4) << 3) | (kkw & 7);
      const int lA0w = ((kkw >> 3) & 1) << 4;
#pragma unroll
      for (int mt = 0; mt < kMT; ++mt) {
#pragma unroll
        for (int r = 0; r < 8; ++r) {
          const int m = mt * 16 + mhalf + r;     // row within 64-row tile
          const float x    = x_lds[m];
          const float hold = h32[m * kH + j];    // (m,j) owned by this thread
          const float rg = sigmoidf_(x * wir + brz + accR[mt][r]);
          const float zg = sigmoidf_(x * wiz + bzz + accZ[mt][r]);
          const float ng = tanhf(x * win + bin + rg * (accN[mt][r] + bhn));
          const float v  = (1.0f - zg) * ng + zg * hold;
          h32[m * kH + j] = v;                   // in-place: thread-private
          Awr[mt * kFragLds + (ktw * 32 + (lA0w + (m & 15))) * 16 + ew] = (__bf16)v;
        }
      }
    }
    __syncthreads();   // h fully updated; A read/write buffers flip next step

    // ---- decoder: y = sigmoid(h . Wout + bout); feed back + emit ----------
    if (dec) {
      const int m0 = tid >> 4, seg = tid & 15;   // m0 in 0..31; rows m0, m0+32
#pragma unroll
      for (int mm = 0; mm < 2; ++mm) {
        const int m = m0 + mm * 32;
        float p = 0.0f;
#pragma unroll 8
        for (int q = 0; q < 32; ++q) {
          const int j = seg * 32 + q;
          p += h32[m * kH + j] * Wout[j];
        }
        part[m][seg] = p;
      }
      __syncthreads();
      if (seg == 0) {
#pragma unroll
        for (int mm = 0; mm < 2; ++mm) {
          const int m = m0 + mm * 32;
          float s = bout[0];
#pragma unroll
          for (int q = 0; q < 16; ++q) s += part[m][q];
          const float y = sigmoidf_(s);
          x_lds[m] = y;
          out[(bbase + m) * kSteps + (step - kS)] = y;
        }
      }
      __syncthreads();
    }
  }
}

// ---------------------------------------------------------------------------
extern "C" void kernel_launch(void* const* d_in, const int* in_sizes, int n_in,
                              void* d_out, int out_size, void* d_ws, size_t ws_size,
                              hipStream_t stream) {
  (void)in_sizes; (void)n_in; (void)out_size; (void)ws_size;
  const float* input_seq = (const float*)d_in[0];
  // d_in[1] = autoregressive_steps (== 32, baked in)
  const float* Wih  = (const float*)d_in[2];
  const float* Whh  = (const float*)d_in[3];
  const float* bih  = (const float*)d_in[4];
  const float* bhh  = (const float*)d_in[5];
  const float* Wout = (const float*)d_in[6];
  const float* bout = (const float*)d_in[7];
  float* out = (float*)d_out;

  __bf16* packed = (__bf16*)d_ws;   // 3H*H bf16 = 1.5 MB scratch

  const int packN = 3 * kH * kH;
  pack_whh_kernel<<<(packN + 255) / 256, 256, 0, stream>>>(Whh, packed);
  gru_main_kernel<<<kB / kBT, kThreads, 0, stream>>>(input_seq, Wih, bih, bhh,
                                                     Wout, bout, packed, out);
}
